// PlanetaryHypergraphMonitor_53523882443005
// MI455X (gfx1250) — matrix operand order repaired and checked by reading, not compiled, not run on hardware
//
#include <hip/hip_runtime.h>
#include <hip/hip_bf16.h>

// ---------------------------------------------------------------------------
// HypergraphConv forward:  mean(relu(D^-1 * H * B^-1 * H^T * (X W) + bias))
// Dominant cost: two nnz-scatter phases (~3.3 GB HBM traffic). GEMM done with
// CDNA5 WMMA bf16 (v_wmma_f32_16x16x32_bf16) — wave32, 16x16x32 tiles.
// ---------------------------------------------------------------------------

#define NUM_NODES      50000
#define NUM_HYPEREDGES 20000
#define IN_DIM         256
#define OUT_DIM        128

typedef __attribute__((ext_vector_type(16))) __bf16 v16bf;
typedef __attribute__((ext_vector_type(8)))  float  v8f;

// --------------------------- degree accumulation ---------------------------
__global__ __launch_bounds__(256)
void degrees_kernel(const int* __restrict__ src, const int* __restrict__ edge,
                    float* __restrict__ deg_n, float* __restrict__ deg_e, int nnz) {
    int i = blockIdx.x * blockDim.x + threadIdx.x;
    if (i < nnz) {
        atomicAdd(&deg_n[src[i]], 1.0f);
        atomicAdd(&deg_e[edge[i]], 1.0f);
    }
}

__global__ __launch_bounds__(256)
void invert_kernel(float* __restrict__ v, int n) {
    int i = blockIdx.x * blockDim.x + threadIdx.x;
    if (i < n) {
        float d = v[i];
        v[i] = (d > 0.0f) ? (1.0f / d) : 0.0f;
    }
}

// ------------------------------ GEMM (WMMA) --------------------------------
// One wave computes one 16x16 tile of xw = X[50000,256] * W[256,128].
// grid = (rows/16, OUT_DIM/16), block = 32 (wave32, EXEC all ones).
__global__ __launch_bounds__(32)
void gemm_wmma_kernel(const float* __restrict__ x, const float* __restrict__ w,
                      float* __restrict__ xw) {
    const int lane = threadIdx.x;        // 0..31
    const int trow = blockIdx.x;         // row tile
    const int tcol = blockIdx.y;         // col tile
    const int m    = lane & 15;
    const int half = lane >> 4;          // 0 or 1

    const int row  = trow * 16 + m;      // A row for this lane
    const int col  = tcol * 16 + m;      // B column for this lane
    const int ka   = half * 8;           // A: K chunks [ka..ka+7], [ka+16..ka+23]
    const int kb   = half * 8;           // B: K chunks [kb..kb+7], [kb+16..kb+23]

    const float* __restrict__ xrow = x + (size_t)row * IN_DIM;

    v8f c = {};
    for (int k0 = 0; k0 < IN_DIM; k0 += 32) {
        v16bf a, b;
        // A fragment: 16-bit A-matrix 16x32 layout (ISA 7.12.2)
#pragma unroll
        for (int i = 0; i < 8; ++i) {
            a[i]     = (__bf16)xrow[k0 + ka + i];
            a[i + 8] = (__bf16)xrow[k0 + ka + 16 + i];
        }
        // B fragment: lane = column, K chunks [kb..kb+7], [kb+16..kb+23] per lane
#pragma unroll
        for (int i = 0; i < 16; ++i) {
            b[i] = (__bf16)w[(size_t)(k0 + kb + (i & 7) + ((i >> 3) << 4)) * OUT_DIM + col];
        }
        c = __builtin_amdgcn_wmma_f32_16x16x32_bf16(
                /*neg_a=*/false, a, /*neg_b=*/false, b,
                /*c_mod=*/(short)0, c, /*reuse_a=*/false, /*reuse_b=*/false);
    }

    // C/D layout: VGPR e -> M = e + 8*half, N = lane&15
    const int nOut  = lane & 15;
    const int mBase = half * 8;
#pragma unroll
    for (int e = 0; e < 8; ++e) {
        xw[(size_t)(trow * 16 + mBase + e) * OUT_DIM + tcol * 16 + nOut] = c[e];
    }
}

// --------------------------- nnz scatter (x2) ------------------------------
// One wave per nnz entry: lane gathers float4 from srcMat[gidx] row and
// atomically accumulates into dstMat[sidx] row (128 floats = 32 lanes x 4).
__global__ __launch_bounds__(256)
void scatter_add_kernel(const float* __restrict__ srcMat,
                        const int* __restrict__ gidx, const int* __restrict__ sidx,
                        float* __restrict__ dstMat, int nnz) {
    int warp = (int)((blockIdx.x * blockDim.x + threadIdx.x) >> 5);
    int lane = threadIdx.x & 31;
    if (warp >= nnz) return;
    int g = gidx[warp];
    int s = sidx[warp];
    const float4 v = *(const float4*)(srcMat + (size_t)g * OUT_DIM + lane * 4);
    float* dst = dstMat + (size_t)s * OUT_DIM + lane * 4;
    atomicAdd(dst + 0, v.x);
    atomicAdd(dst + 1, v.y);
    atomicAdd(dst + 2, v.z);
    atomicAdd(dst + 3, v.w);
}

__global__ __launch_bounds__(256)
void scale_rows_kernel(float* __restrict__ mat, const float* __restrict__ inv, int total) {
    int i = blockIdx.x * blockDim.x + threadIdx.x;
    if (i < total) mat[i] *= inv[i >> 7];   // OUT_DIM == 128
}

// ------------------------- finalize + mean pool ----------------------------
// block = 128 threads (one feature each); each block reduces 256 nodes in
// registers, then one atomicAdd per feature.
__global__ __launch_bounds__(128)
void finalize_mean_kernel(const float* __restrict__ out_acc,
                          const float* __restrict__ d_inv,
                          const float* __restrict__ bias,
                          float* __restrict__ acc, int num_nodes) {
    int f  = threadIdx.x;
    int n0 = blockIdx.x * 256;
    int n1 = n0 + 256; if (n1 > num_nodes) n1 = num_nodes;
    float b = bias[f];
    float s = 0.0f;
    for (int n = n0; n < n1; ++n) {
        float v = out_acc[(size_t)n * OUT_DIM + f] * d_inv[n] + b;
        s += (v > 0.0f) ? v : 0.0f;
    }
    atomicAdd(&acc[f], s);
}

__global__ __launch_bounds__(128)
void write_out_kernel(const float* __restrict__ acc, float* __restrict__ out, int num_nodes) {
    int f = threadIdx.x;
    out[f] = acc[f] / (float)num_nodes;
}

// ------------------------------- launcher ----------------------------------
static inline size_t align256(size_t x) { return (x + 255) & ~(size_t)255; }

extern "C" void kernel_launch(void* const* d_in, const int* in_sizes, int n_in,
                              void* d_out, int out_size, void* d_ws, size_t ws_size,
                              hipStream_t stream) {
    const float* x    = (const float*)d_in[0];   // [N, 256]
    const float* w    = (const float*)d_in[1];   // [256, 128]
    const float* bias = (const float*)d_in[2];   // [128]
    const int*   hei  = (const int*)d_in[3];     // [2, nnz]

    const int nnz  = in_sizes[3] / 2;
    const int N    = NUM_NODES;
    const int E    = NUM_HYPEREDGES;
    const int* src  = hei;        // node ids
    const int* edge = hei + nnz;  // hyperedge ids

    // workspace carve-out
    char* p = (char*)d_ws;
    float* xw      = (float*)p; p += align256((size_t)N * OUT_DIM * sizeof(float));
    float* m_e     = (float*)p; p += align256((size_t)E * OUT_DIM * sizeof(float));
    float* out_acc = (float*)p; p += align256((size_t)N * OUT_DIM * sizeof(float));
    float* deg_n   = (float*)p; p += align256((size_t)N * sizeof(float));
    float* deg_e   = (float*)p; p += align256((size_t)E * sizeof(float));
    float* acc     = (float*)p; p += align256((size_t)OUT_DIM * sizeof(float));
    (void)ws_size;

    // zero the accumulation buffers (graph-capturable async memsets)
    hipMemsetAsync(m_e,     0, (size_t)E * OUT_DIM * sizeof(float), stream);
    hipMemsetAsync(out_acc, 0, (size_t)N * OUT_DIM * sizeof(float), stream);
    hipMemsetAsync(deg_n,   0, (size_t)N * sizeof(float), stream);
    hipMemsetAsync(deg_e,   0, (size_t)E * sizeof(float), stream);
    hipMemsetAsync(acc,     0, (size_t)OUT_DIM * sizeof(float), stream);

    // 1) degrees, then invert in place -> d_inv / b_inv
    degrees_kernel<<<(nnz + 255) / 256, 256, 0, stream>>>(src, edge, deg_n, deg_e, nnz);
    invert_kernel<<<(N + 255) / 256, 256, 0, stream>>>(deg_n, N);
    invert_kernel<<<(E + 255) / 256, 256, 0, stream>>>(deg_e, E);

    // 2) xw = X W  (WMMA bf16, one wave per 16x16 tile; N = 3125*16 exactly)
    dim3 ggrid(N / 16, OUT_DIM / 16);
    gemm_wmma_kernel<<<ggrid, 32, 0, stream>>>(x, w, xw);

    // 3) node -> hyperedge scatter:  m_e[edge] += xw[src]
    {
        long long warps = nnz;
        long long threads = warps * 32;
        scatter_add_kernel<<<(unsigned)((threads + 255) / 256), 256, 0, stream>>>(
            xw, src, edge, m_e, nnz);
    }
    // 4) m_e *= b_inv
    scale_rows_kernel<<<((E * OUT_DIM) + 255) / 256, 256, 0, stream>>>(m_e, deg_e, E * OUT_DIM);

    // 5) hyperedge -> node scatter:  out_acc[src] += m_e[edge]
    {
        long long warps = nnz;
        long long threads = warps * 32;
        scatter_add_kernel<<<(unsigned)((threads + 255) / 256), 256, 0, stream>>>(
            m_e, edge, src, out_acc, nnz);
    }

    // 6) relu(out_acc * d_inv + bias) reduced over nodes, then mean
    finalize_mean_kernel<<<(N + 255) / 256, 128, 0, stream>>>(out_acc, deg_n, bias, acc, N);
    write_out_kernel<<<1, 128, 0, stream>>>(acc, (float*)d_out, N);
}
